// EGNNLayer_8684423872621
// MI455X (gfx1250) — matrix operand order repaired and checked by reading, not compiled, not run on hardware
//
#include <hip/hip_runtime.h>
#include <hip/hip_bf16.h>

#define Hdim 128
#define WPB 8   // waves per 256-thread block (wave32)

typedef __bf16 bf16_t;
typedef bf16_t bf16x16 __attribute__((ext_vector_type(16)));
typedef float  f32x8   __attribute__((ext_vector_type(8)));
typedef unsigned int u32x4 __attribute__((ext_vector_type(4)));

union Frag { bf16x16 v; u32x4 q[2]; };

__device__ __forceinline__ float silu_f(float x) {
    return x * (1.0f / (1.0f + __expf(-x)));
}

__device__ __forceinline__ f32x8 wmma_bf16(const Frag& a, const Frag& b, f32x8 c) {
    // D(16x16,f32) = A(16x32,bf16) * B(32x16,bf16) + C
    return __builtin_amdgcn_wmma_f32_16x16x32_bf16(false, a.v, false, b.v,
                                                   (short)0, c, false, false);
}

__device__ __forceinline__ void atomAddF(float* p, float v) {
    unsafeAtomicAdd(p, v);   // global_atomic_add_f32
}

// ---------------------------------------------------------------------------
// Prep kernels
// ---------------------------------------------------------------------------
__global__ void zero_f32(float* __restrict__ p, size_t n) {
    size_t i = (size_t)blockIdx.x * blockDim.x + threadIdx.x;
    size_t st = (size_t)gridDim.x * blockDim.x;
    for (; i < n; i += st) p[i] = 0.0f;
}

__global__ void cvt_bf16(const float* __restrict__ s, bf16_t* __restrict__ d, size_t n) {
    size_t i = (size_t)blockIdx.x * blockDim.x + threadIdx.x;
    size_t st = (size_t)gridDim.x * blockDim.x;
    for (; i < n; i += st) d[i] = (bf16_t)s[i];
}

// Pack W[K][128] (row-major f32) into per-lane bf16 B-fragment blocks.
// Block = (ktile, ntile): 32 lanes x 16 bf16. Lane<16: K=kt*32+{0..7,16..23} of col
// N=nt*16+lane; lane>=16: K=kt*32+{8..15,24..31} of col N=nt*16+(lane-16).
// gridDim.x = ktiles*ntiles, blockDim.x = 32.
__global__ void pack_weights(const float* __restrict__ W, int ntiles,
                             bf16_t* __restrict__ dst) {
    int blk  = blockIdx.x;
    int lane = threadIdx.x;
    int kt = blk / ntiles, nt = blk % ntiles;
    int kbase = kt * 32 + ((lane >> 4) << 3);
    int n     = nt * 16 + (lane & 15);
    bf16_t* o = dst + ((size_t)blk * 32 + lane) * 16;
#pragma unroll
    for (int j = 0; j < 16; ++j)
        o[j] = (bf16_t)W[(size_t)(kbase + j + ((j >> 3) << 3)) * Hdim + n];
}

// 128-K GEMM accumulate: acc[8] (16x128 f32 D tiles) += A(16x128) * pW.
// arow points at this lane's A row (row m of the tile), contiguous bf16[128].
__device__ __forceinline__ void mm128(f32x8 acc[8], const bf16_t* arow,
                                      const bf16_t* __restrict__ pW,
                                      int lane, int half) {
#pragma unroll
    for (int ks = 0; ks < 4; ++ks) {
        Frag a;
        int base = ks * 32 + (half << 3);
        a.q[0] = *(const u32x4*)(arow + base);
        a.q[1] = *(const u32x4*)(arow + base + 16);
#pragma unroll
        for (int t = 0; t < 8; ++t) {
            Frag b;
            const bf16_t* wp = pW + (((size_t)(ks * 8 + t)) * 32 + lane) * 16;
            b.q[0] = *(const u32x4*)(wp);
            b.q[1] = *(const u32x4*)(wp + 8);
            acc[t] = wmma_bf16(a, b, acc[t]);
        }
    }
}

// ---------------------------------------------------------------------------
// Edge kernel: one wave32 per 16-edge tile.
// ---------------------------------------------------------------------------
__global__ void __launch_bounds__(256) egnn_edge_kernel(
    const float* __restrict__ x, const int* __restrict__ ei,
    const float* __restrict__ We1, const float* __restrict__ be1,
    const float* __restrict__ be2, const float* __restrict__ bc1,
    const float* __restrict__ Wc2, const float* __restrict__ bc2,
    const bf16_t* __restrict__ hbf,
    const bf16_t* __restrict__ pWe1a, const bf16_t* __restrict__ pWe1b,
    const bf16_t* __restrict__ pWe2,  const bf16_t* __restrict__ pWc1,
    float* __restrict__ counts, float* __restrict__ dxacc,
    float* __restrict__ aggacc, int E)
{
    __shared__ bf16_t s_ef[WPB][16 * Hdim];
    __shared__ int    s_row[WPB][16];
    __shared__ int    s_col[WPB][16];
    __shared__ float  s_d2[WPB][16];
    __shared__ float  s_rel[WPB][16][3];
    __shared__ float  s_gate[WPB][16];

    const int wave = threadIdx.x >> 5;
    const int lane = threadIdx.x & 31;
    const int tile = blockIdx.x * WPB + wave;
    const int e0   = tile << 4;
    if (e0 >= E) return;                       // wave-uniform
    const int m    = lane & 15;
    const int half = lane >> 4;
    const int off8 = half << 3;

    // per-edge geometry (lanes 0..15, one edge each)
    if (lane < 16) {
        int e = e0 + m;
        int r = 0, c = 0;
        float d2 = 0.f, r0 = 0.f, r1 = 0.f, r2 = 0.f;
        if (e < E) {
            r = ei[e];
            c = ei[E + e];
            r0 = x[r * 3 + 0] - x[c * 3 + 0];
            r1 = x[r * 3 + 1] - x[c * 3 + 1];
            r2 = x[r * 3 + 2] - x[c * 3 + 2];
            d2 = r0 * r0 + r1 * r1 + r2 * r2;
        }
        s_row[wave][m] = r;  s_col[wave][m] = c;  s_d2[wave][m] = d2;
        s_rel[wave][m][0] = r0; s_rel[wave][m][1] = r1; s_rel[wave][m][2] = r2;
    }

    // ---- stage 1: ef1 = silu([h_row | h_col | d2] @ We1 + be1)
    // d2 * We1[256,:] + be1 folded into the f32 accumulator init (exact).
    float d2v[8];
#pragma unroll
    for (int v = 0; v < 8; ++v) d2v[v] = s_d2[wave][off8 + v];

    f32x8 acc[8];
#pragma unroll
    for (int t = 0; t < 8; ++t) {
        int cc = t * 16 + m;
        float b  = be1[cc];
        float wl = We1[256 * Hdim + cc];
#pragma unroll
        for (int v = 0; v < 8; ++v) acc[t][v] = __builtin_fmaf(d2v[v], wl, b);
    }

    const bf16_t* hr = hbf + (size_t)s_row[wave][m] * Hdim;
    const bf16_t* hc = hbf + (size_t)s_col[wave][m] * Hdim;
    mm128(acc, hr, pWe1a, lane, half);   // h[row] @ We1[0:128]
    mm128(acc, hc, pWe1b, lane, half);   // h[col] @ We1[128:256]

#pragma unroll
    for (int t = 0; t < 8; ++t)
#pragma unroll
        for (int v = 0; v < 8; ++v)
            s_ef[wave][(off8 + v) * Hdim + t * 16 + m] = (bf16_t)silu_f(acc[t][v]);

    // ---- stage 2: ef = silu(ef1 @ We2 + be2)
#pragma unroll
    for (int t = 0; t < 8; ++t) {
        float b = be2[t * 16 + m];
#pragma unroll
        for (int v = 0; v < 8; ++v) acc[t][v] = b;
    }
    mm128(acc, &s_ef[wave][m * Hdim], pWe2, lane, half);
#pragma unroll
    for (int t = 0; t < 8; ++t)
#pragma unroll
        for (int v = 0; v < 8; ++v)
            s_ef[wave][(off8 + v) * Hdim + t * 16 + m] = (bf16_t)silu_f(acc[t][v]);

    // ---- stage 3: g1 = ef @ Wc1 + bc1 (pre-silu kept in regs)
#pragma unroll
    for (int t = 0; t < 8; ++t) {
        float b = bc1[t * 16 + m];
#pragma unroll
        for (int v = 0; v < 8; ++v) acc[t][v] = b;
    }
    mm128(acc, &s_ef[wave][m * Hdim], pWc1, lane, half);

    // gate[e] = silu(g1) @ Wc2 + bc2 : xor-shuffle reduce over 16-lane halves
    float wc2v[8];
#pragma unroll
    for (int t = 0; t < 8; ++t) wc2v[t] = Wc2[t * 16 + m];
    float bc2s = bc2[0];
#pragma unroll
    for (int v = 0; v < 8; ++v) {
        float p = 0.f;
#pragma unroll
        for (int t = 0; t < 8; ++t) p += silu_f(acc[t][v]) * wc2v[t];
        p += __shfl_xor(p, 1, 32);
        p += __shfl_xor(p, 2, 32);
        p += __shfl_xor(p, 4, 32);
        p += __shfl_xor(p, 8, 32);          // masks<16 keep halves separate
        if (m == v) s_gate[wave][off8 + v] = p + bc2s;
    }

    // ---- scatter-mean accumulation (f32 L2 atomics)
    if (lane < 16 && (e0 + m) < E) {
        int   r = s_row[wave][m];
        float g = s_gate[wave][m];
        atomAddF(&counts[r], 1.0f);
        atomAddF(&dxacc[(size_t)r * 3 + 0], s_rel[wave][m][0] * g);
        atomAddF(&dxacc[(size_t)r * 3 + 1], s_rel[wave][m][1] * g);
        atomAddF(&dxacc[(size_t)r * 3 + 2], s_rel[wave][m][2] * g);
    }
#pragma unroll 4
    for (int i = lane; i < 16 * Hdim; i += 32) {
        int rr = i >> 7;                     // edge-in-tile
        if (e0 + rr < E)
            atomAddF(&aggacc[(size_t)s_row[wave][rr] * Hdim + (i & (Hdim - 1))],
                     (float)s_ef[wave][i]);
    }
}

// ---------------------------------------------------------------------------
// Node kernel: one wave32 per 16-node tile.
// ---------------------------------------------------------------------------
__global__ void __launch_bounds__(256) egnn_node_kernel(
    const float* __restrict__ x, const float* __restrict__ h,
    const float* __restrict__ bn1, const float* __restrict__ bn2,
    const float* __restrict__ gamma, const float* __restrict__ beta,
    const bf16_t* __restrict__ hbf,
    const bf16_t* __restrict__ pWn1, const bf16_t* __restrict__ pWn2,
    const float* __restrict__ counts, const float* __restrict__ dxacc,
    const float* __restrict__ aggacc,
    float* __restrict__ xout, float* __restrict__ hout, int N)
{
    __shared__ bf16_t s_t[WPB][16 * Hdim];
    __shared__ int    s_node[WPB][16];
    __shared__ float  s_inv[WPB][16];

    const int wave = threadIdx.x >> 5;
    const int lane = threadIdx.x & 31;
    const int tile = blockIdx.x * WPB + wave;
    if ((tile << 4) >= N) return;            // wave-uniform
    const int m    = lane & 15;
    const int half = lane >> 4;
    const int off8 = half << 3;

    if (lane < 16) {
        int  idx = tile * 16 + m;
        bool val = idx < N;
        int  nd  = val ? idx : (N - 1);
        float inv = 1.0f / fmaxf(counts[nd], 1.0f);
        s_node[wave][m] = nd;
        s_inv[wave][m]  = inv;
        if (val) {
#pragma unroll
            for (int c = 0; c < 3; ++c)
                xout[(size_t)nd * 3 + c] =
                    x[(size_t)nd * 3 + c] + dxacc[(size_t)nd * 3 + c] * inv;
        }
    }

    // ---- n1 = silu([h | agg/cnt] @ Wn1 + bn1)
    f32x8 acc[8];
#pragma unroll
    for (int t = 0; t < 8; ++t) {
        float b = bn1[t * 16 + m];
#pragma unroll
        for (int v = 0; v < 8; ++v) acc[t][v] = b;
    }
    const int    nd  = s_node[wave][m];
    const float  inv = s_inv[wave][m];
    mm128(acc, hbf + (size_t)nd * Hdim, pWn1, lane, half);  // h half (K 0..127)

    const float* ap = aggacc + (size_t)nd * Hdim;           // agg half (K 128..255)
#pragma unroll
    for (int ks = 0; ks < 4; ++ks) {
        Frag a;
        int base = ks * 32 + (half << 3);
#pragma unroll
        for (int j = 0; j < 8; ++j) {
            a.v[j]     = (bf16_t)(ap[base + j]      * inv);
            a.v[8 + j] = (bf16_t)(ap[base + 16 + j] * inv);
        }
#pragma unroll
        for (int t = 0; t < 8; ++t) {
            Frag b;
            const bf16_t* wp = pWn1 + 16384 + (((size_t)(ks * 8 + t)) * 32 + lane) * 16;
            b.q[0] = *(const u32x4*)(wp);
            b.q[1] = *(const u32x4*)(wp + 8);
            acc[t] = wmma_bf16(a, b, acc[t]);
        }
    }
#pragma unroll
    for (int t = 0; t < 8; ++t)
#pragma unroll
        for (int v = 0; v < 8; ++v)
            s_t[wave][(off8 + v) * Hdim + t * 16 + m] = (bf16_t)silu_f(acc[t][v]);

    // ---- dh = n1 @ Wn2 + bn2 ; h2 = h + dh (f32 residual)
#pragma unroll
    for (int t = 0; t < 8; ++t) {
        float b = bn2[t * 16 + m];
#pragma unroll
        for (int v = 0; v < 8; ++v) acc[t][v] = b;
    }
    mm128(acc, &s_t[wave][m * Hdim], pWn2, lane, half);

    int ndv[8];
#pragma unroll
    for (int v = 0; v < 8; ++v) ndv[v] = s_node[wave][off8 + v];
#pragma unroll
    for (int t = 0; t < 8; ++t)
#pragma unroll
        for (int v = 0; v < 8; ++v)
            acc[t][v] += h[(size_t)ndv[v] * Hdim + t * 16 + m];

    // ---- LayerNorm over 128 cols (8 per lane x 16 lanes/half) + silu
    float gam[8], bet[8];
#pragma unroll
    for (int t = 0; t < 8; ++t) { gam[t] = gamma[t * 16 + m]; bet[t] = beta[t * 16 + m]; }

#pragma unroll
    for (int v = 0; v < 8; ++v) {
        float sum = 0.f, sq = 0.f;
#pragma unroll
        for (int t = 0; t < 8; ++t) { float a = acc[t][v]; sum += a; sq += a * a; }
        sum += __shfl_xor(sum, 1, 32); sq += __shfl_xor(sq, 1, 32);
        sum += __shfl_xor(sum, 2, 32); sq += __shfl_xor(sq, 2, 32);
        sum += __shfl_xor(sum, 4, 32); sq += __shfl_xor(sq, 4, 32);
        sum += __shfl_xor(sum, 8, 32); sq += __shfl_xor(sq, 8, 32);
        float mean = sum * (1.0f / 128.0f);
        float var  = sq * (1.0f / 128.0f) - mean * mean;
        float rstd = rsqrtf(var + 1e-5f);
        bool  val  = (tile * 16 + off8 + v) < N;
        int   nr   = ndv[v];
#pragma unroll
        for (int t = 0; t < 8; ++t) {
            float o = silu_f((acc[t][v] - mean) * rstd * gam[t] + bet[t]);
            if (val) hout[(size_t)nr * Hdim + t * 16 + m] = o;
        }
    }
}

// ---------------------------------------------------------------------------
// Host launcher
// ---------------------------------------------------------------------------
extern "C" void kernel_launch(void* const* d_in, const int* in_sizes, int n_in,
                              void* d_out, int out_size, void* d_ws, size_t ws_size,
                              hipStream_t stream)
{
    const float* x    = (const float*)d_in[0];
    const float* h    = (const float*)d_in[1];
    const int*   ei   = (const int*)  d_in[2];
    const float* We1  = (const float*)d_in[3];
    const float* be1  = (const float*)d_in[4];
    const float* We2  = (const float*)d_in[5];
    const float* be2  = (const float*)d_in[6];
    const float* Wc1  = (const float*)d_in[7];
    const float* bc1  = (const float*)d_in[8];
    const float* Wc2  = (const float*)d_in[9];
    const float* bc2  = (const float*)d_in[10];
    const float* Wn1  = (const float*)d_in[11];
    const float* bn1  = (const float*)d_in[12];
    const float* Wn2  = (const float*)d_in[13];
    const float* bn2  = (const float*)d_in[14];
    const float* gamm = (const float*)d_in[15];
    const float* beta = (const float*)d_in[16];

    const int N = in_sizes[0] / 3;
    const int E = in_sizes[2] / 2;

    char* ws = (char*)d_ws;
    auto al = [](size_t v) { return (v + 255) & ~(size_t)255; };
    size_t off = 0;
    float*  counts = (float*)(ws + off);  off = al(off + (size_t)N * 4);
    float*  dxacc  = (float*)(ws + off);  off = al(off + (size_t)N * 3 * 4);
    float*  aggacc = (float*)(ws + off);
    size_t zeroBytes = off + (size_t)N * Hdim * 4;  off = al(zeroBytes);
    bf16_t* hbf    = (bf16_t*)(ws + off); off = al(off + (size_t)N * Hdim * 2);
    bf16_t* pWe1a  = (bf16_t*)(ws + off); off += 32 * 1024;
    bf16_t* pWe1b  = (bf16_t*)(ws + off); off += 32 * 1024;
    bf16_t* pWe2   = (bf16_t*)(ws + off); off += 32 * 1024;
    bf16_t* pWc1   = (bf16_t*)(ws + off); off += 32 * 1024;
    bf16_t* pWn1   = (bf16_t*)(ws + off); off += 64 * 1024;
    bf16_t* pWn2   = (bf16_t*)(ws + off); off += 32 * 1024;

    zero_f32<<<1024, 256, 0, stream>>>((float*)ws, zeroBytes / 4);
    cvt_bf16<<<1024, 256, 0, stream>>>(h, hbf, (size_t)N * Hdim);
    pack_weights<<<32, 32, 0, stream>>>(We1, 8, pWe1a);                 // K rows 0..127
    pack_weights<<<32, 32, 0, stream>>>(We1 + 128 * Hdim, 8, pWe1b);    // K rows 128..255
    pack_weights<<<32, 32, 0, stream>>>(We2, 8, pWe2);
    pack_weights<<<32, 32, 0, stream>>>(Wc1, 8, pWc1);
    pack_weights<<<64, 32, 0, stream>>>(Wn1, 8, pWn1);                  // K=256
    pack_weights<<<32, 32, 0, stream>>>(Wn2, 8, pWn2);

    const int tilesE = (E + 15) / 16;
    egnn_edge_kernel<<<(tilesE + WPB - 1) / WPB, 256, 0, stream>>>(
        x, ei, We1, be1, be2, bc1, Wc2, bc2,
        hbf, pWe1a, pWe1b, pWe2, pWc1,
        counts, dxacc, aggacc, E);

    float* xout = (float*)d_out;
    float* hout = xout + (size_t)N * 3;
    const int tilesN = (N + 15) / 16;
    egnn_node_kernel<<<(tilesN + WPB - 1) / WPB, 256, 0, stream>>>(
        x, h, bn1, bn2, gamm, beta,
        hbf, pWn1, pWn2, counts, dxacc, aggacc,
        xout, hout, N);
}
